// SSGNModel_83494164234358
// MI455X (gfx1250) — compile-verified, matrix operand order checked by reading, not run on hardware
//
#include <hip/hip_runtime.h>
#include <hip/hip_bf16.h>

// ---------------- problem constants (from reference) ----------------
#define NN   10000
#define EE   160000
#define GG   64
#define FIN  128
#define HH   256
#define NEG_INF_MAPPED 0x007FFFFFu   // fmap(-inf)

typedef __attribute__((ext_vector_type(16))) __bf16 v16bf;
typedef __attribute__((ext_vector_type(8)))  __bf16 v8bf;
typedef __attribute__((ext_vector_type(4)))  __bf16 v4bf;
typedef __attribute__((ext_vector_type(8)))  float  v8f;

#define CDIV(a,b) (((a)+(b)-1)/(b))

// ---------------- device helpers ----------------
__device__ __forceinline__ unsigned fmap(float x) {
    unsigned u = __float_as_uint(x);
    return (u & 0x80000000u) ? ~u : (u | 0x80000000u);
}
__device__ __forceinline__ float funmap(unsigned u) {
    unsigned v = (u & 0x80000000u) ? (u & 0x7FFFFFFFu) : ~u;
    return __uint_as_float(v);
}
__device__ __forceinline__ __bf16 tobf(float f) {
    unsigned u = __float_as_uint(f);
    unsigned r = u + 0x7FFFu + ((u >> 16) & 1u);   // round-to-nearest-even
    unsigned short h = (unsigned short)(r >> 16);
    __bf16 b;
    __builtin_memcpy(&b, &h, 2);
    return b;
}

// ---------------- utility kernels ----------------
__global__ void fill_f32(float* p, float v, int n) {
    int i = blockIdx.x * blockDim.x + threadIdx.x;
    if (i < n) p[i] = v;
}
__global__ void fill_u32(unsigned* p, unsigned v, int n) {
    int i = blockIdx.x * blockDim.x + threadIdx.x;
    if (i < n) p[i] = v;
}

// degree accumulation (float adds: counts < 2^24, exact)
__global__ void degrees_k(const int* __restrict__ src, const int* __restrict__ dst,
                          float* dsrc, float* ddst) {
    int e = blockIdx.x * blockDim.x + threadIdx.x;
    if (e >= EE) return;
    atomicAdd(&dsrc[src[e]], 1.0f);
    atomicAdd(&ddst[dst[e]], 1.0f);
}
__global__ void deg_finalize_k(float* p, int n) {   // p = clip(deg,1)^-0.5
    int i = blockIdx.x * blockDim.x + threadIdx.x;
    if (i < n) p[i] = rsqrtf(fmaxf(p[i], 1.0f));
}
__global__ void counts_k(const int* __restrict__ gid, float* cnt) {
    int i = blockIdx.x * blockDim.x + threadIdx.x;
    if (i < NN) atomicAdd(&cnt[gid[i]], 1.0f);
}

// W[K x Hout] (row-major) -> Wt[Hout x K] (row-major, bf16)
__global__ void wtrans_k(const float* __restrict__ W, __bf16* __restrict__ Wt,
                         int K, int Hout) {
    int i = blockIdx.x * blockDim.x + threadIdx.x;
    if (i >= K * Hout) return;
    int c = i % Hout, k = i / Hout;
    Wt[(size_t)c * K + k] = tobf(W[(size_t)k * Hout + c]);
}

// activation f32 -> bf16 (optional per-row scale fused); 4 elems/thread
__global__ void cvt_bf16_k(const float* __restrict__ X, const float* __restrict__ rowScale,
                           __bf16* __restrict__ Y, int total4, int kshift) {
    int idx = blockIdx.x * blockDim.x + threadIdx.x;
    if (idx >= total4) return;
    int f = idx << 2;
    float sc = rowScale ? rowScale[f >> kshift] : 1.0f;
    const float4 x = *(const float4*)(X + f);
    v4bf y;
    y[0] = tobf(x.x * sc);
    y[1] = tobf(x.y * sc);
    y[2] = tobf(x.z * sc);
    y[3] = tobf(x.w * sc);
    *(v4bf*)(Y + f) = y;
}

// ---------------- GCN scatter: AGG[dst,:] += X[src,:] * d_src[src] ----------------
__global__ void gcn_scatter_k(const float* __restrict__ X, const int* __restrict__ src,
                              const int* __restrict__ dst, const float* __restrict__ dsrc,
                              float* __restrict__ AGG, int K) {
    int idx = blockIdx.x * blockDim.x + threadIdx.x;
    int chunks = K >> 2;
    if (idx >= EE * chunks) return;
    int e = idx / chunks;
    int f = (idx - e * chunks) << 2;
    int s = src[e], d = dst[e];
    float sc = dsrc[s];
    const float4 x = *(const float4*)(X + (size_t)s * K + f);
    float* o = AGG + (size_t)d * K + f;
    atomicAdd(o + 0, x.x * sc);
    atomicAdd(o + 1, x.y * sc);
    atomicAdd(o + 2, x.z * sc);
    atomicAdd(o + 3, x.w * sc);
}

// ---------------- WMMA GEMM: C = act(Abf @ Wt^T + bias) ----------------
// Abf: Nrows x K bf16 row-major (pre-converted, row-scaled activations).
// Wt:  Hout x K bf16 row-major (pre-transposed weight).
// One 16x16 tile per wave; block = 4 waves = 16 rows x 64 cols.
__global__ void gemm_wmma_k(const __bf16* __restrict__ Abf, const __bf16* __restrict__ Wt,
                            const float* __restrict__ bias, float* __restrict__ C,
                            int Nrows, int K, int Hout, int doRelu) {
    int wave = threadIdx.x >> 5;
    int lane = threadIdx.x & 31;
    int half = lane >> 4;
    int l    = lane & 15;

    int rowBase = blockIdx.x * 16;
    int colBase = blockIdx.y * 64 + wave * 16;

    int aRow = rowBase + l;
    // Clamp instead of predicating: an invalid A lane only feeds C rows that
    // are never stored, so loads can be unconditional wide b128s.
    int aRowC = (aRow < Nrows) ? aRow : (Nrows - 1);
    int bCol = colBase + l;                     // Hout is a multiple of 64

    const __bf16* arow = Abf + (size_t)aRowC * K;
    const __bf16* brow = Wt  + (size_t)bCol  * K;

    union frag { v16bf v; v8bf h[2]; };

    v8f acc = {};
#pragma unroll 4
    for (int kb = 0; kb < K; kb += 32) {
        if (kb + 32 < K)
            __builtin_prefetch(arow + kb + 32, 0, 1);   // global_prefetch_b8
        int k0 = kb + half * 8;
        frag a, b;
        a.h[0] = *(const v8bf*)(arow + k0);        // K = k0..k0+7      (16B)
        a.h[1] = *(const v8bf*)(arow + k0 + 16);   // K = k0+16..k0+23  (16B)
        b.h[0] = *(const v8bf*)(brow + k0);
        b.h[1] = *(const v8bf*)(brow + k0 + 16);
        acc = __builtin_amdgcn_wmma_f32_16x16x32_bf16(
            /*neg_a=*/false, a.v, /*neg_b=*/false, b.v,
            /*c_mod=*/(short)0, acc, /*reuse_a=*/false, /*reuse_b=*/false);
    }

    float bv = bias ? bias[bCol] : 0.0f;
#pragma unroll
    for (int v = 0; v < 8; ++v) {
        int r = rowBase + v + half * 8;
        if (r < Nrows) {
            float x = acc[v] + bv;
            if (doRelu) x = fmaxf(x, 0.0f);
            C[(size_t)r * Hout + bCol] = x;
        }
    }
}

// ---------------- GAT pieces ----------------
// EL[i,h] = dot(Z[i, h*H : h*H+H], al[h,:]); ER likewise. One wave per (i,h).
__global__ void gat_dots_k(const float* __restrict__ Z, const float* __restrict__ al,
                           const float* __restrict__ ar, float* __restrict__ EL,
                           float* __restrict__ ER) {
    int wid  = blockIdx.x * (blockDim.x >> 5) + (threadIdx.x >> 5);
    int lane = threadIdx.x & 31;
    if (wid >= NN * 2) return;
    int i = wid >> 1, h = wid & 1;
    const float* z = Z + (size_t)i * (2 * HH) + h * HH;
    const float* a = al + h * HH;
    const float* r = ar + h * HH;
    float sl = 0.0f, sr = 0.0f;
#pragma unroll
    for (int t = 0; t < HH / 32; ++t) {
        float zv = z[t * 32 + lane];
        sl += zv * a[t * 32 + lane];
        sr += zv * r[t * 32 + lane];
    }
#pragma unroll
    for (int m = 16; m >= 1; m >>= 1) {
        sl += __shfl_xor(sl, m, 32);
        sr += __shfl_xor(sr, m, 32);
    }
    if (lane == 0) { EL[wid] = sl; ER[wid] = sr; }
}

__global__ void edge_max_k(const int* __restrict__ src, const int* __restrict__ dst,
                           const float* __restrict__ EL, const float* __restrict__ ER,
                           float* __restrict__ EVAL, unsigned* __restrict__ EMAX) {
    int idx = blockIdx.x * blockDim.x + threadIdx.x;
    if (idx >= EE * 2) return;
    int e = idx >> 1, h = idx & 1;
    int s = src[e], d = dst[e];
    float m = EL[s * 2 + h] + ER[d * 2 + h];
    m = (m > 0.0f) ? m : 0.2f * m;           // leaky_relu(., 0.2)
    EVAL[idx] = m;
    atomicMax(&EMAX[d * 2 + h], fmap(m));
}

__global__ void edge_exp_k(const int* __restrict__ dst, float* __restrict__ EVAL,
                           const unsigned* __restrict__ EMAX, float* __restrict__ EDEN) {
    int idx = blockIdx.x * blockDim.x + threadIdx.x;
    if (idx >= EE * 2) return;
    int e = idx >> 1, h = idx & 1;
    int d = dst[e];
    float ex = __expf(EVAL[idx] - funmap(EMAX[d * 2 + h]));
    EVAL[idx] = ex;
    atomicAdd(&EDEN[d * 2 + h], ex);
}

// O[dst, h*H+f] += alpha * Z[src, h*H+f]
__global__ void edge_aggr_k(const int* __restrict__ src, const int* __restrict__ dst,
                            const float* __restrict__ Z, const float* __restrict__ EVAL,
                            const float* __restrict__ EDEN, float* __restrict__ O) {
    int idx = blockIdx.x * blockDim.x + threadIdx.x;       // EE * 2 * (HH/4)
    const int per_edge = 2 * (HH / 4);
    if (idx >= EE * per_edge) return;
    int e = idx / per_edge;
    int r = idx - e * per_edge;
    int h = r >> 6;
    int f = (r & 63) << 2;
    int s = src[e], d = dst[e];
    float alpha = EVAL[e * 2 + h] / fmaxf(EDEN[d * 2 + h], 1e-20f);
    const float4 z = *(const float4*)(Z + (size_t)s * (2 * HH) + h * HH + f);
    float* o = O + (size_t)d * (2 * HH) + h * HH + f;
    atomicAdd(o + 0, alpha * z.x);
    atomicAdd(o + 1, alpha * z.y);
    atomicAdd(o + 2, alpha * z.z);
    atomicAdd(o + 3, alpha * z.w);
}

// out[i,f] = mean_h relu(O[i,h*H+f] + b[h*H+f])
__global__ void gat_final_k(const float* __restrict__ O, const float* __restrict__ b,
                            float* __restrict__ out) {
    int idx = blockIdx.x * blockDim.x + threadIdx.x;
    if (idx >= NN * HH) return;
    int i = idx >> 8, f = idx & (HH - 1);
    float a0 = fmaxf(O[(size_t)i * (2 * HH) + f]      + b[f],      0.0f);
    float a1 = fmaxf(O[(size_t)i * (2 * HH) + HH + f] + b[HH + f], 0.0f);
    out[idx] = 0.5f * (a0 + a1);
}

// ---------------- readout accumulation ----------------
__global__ void readout_k(const float* __restrict__ X, const int* __restrict__ gid,
                          float* __restrict__ avg, unsigned* __restrict__ mx) {
    int idx = blockIdx.x * blockDim.x + threadIdx.x;
    if (idx >= NN * HH) return;
    int i = idx >> 8, f = idx & (HH - 1);
    int g = gid[i];
    float x = X[idx];
    atomicAdd(&avg[g * HH + f], x);
    atomicMax(&mx[g * HH + f], fmap(x));
}

// ---------------- merged / MLP ----------------
// RA/RM layout: [5][G*H]; 0=gcn1 1=gcn2 2=gcn3 3=g2 4=g3
__global__ void merged_k(const float* __restrict__ RA, const unsigned* __restrict__ RM,
                         const float* __restrict__ cnt, float* __restrict__ M) {
    int idx = blockIdx.x * blockDim.x + threadIdx.x;
    if (idx >= GG * 1024) return;
    int g = idx >> 10, j = idx & 1023;
    float c = fmaxf(cnt[g], 1.0f);
    const int S = GG * HH;
    float v;
    if (j < 512) {                               // spec = r1 + r2 + r3
        int f = j & (HH - 1);
        int o = g * HH + f;
        if (j < HH)
            v = (RA[o] + RA[S + o] + RA[2 * S + o]) / c;
        else
            v = funmap(RM[o]) + funmap(RM[S + o]) + funmap(RM[2 * S + o]);
    } else {                                     // spat = r1 + s2 + s3
        int jj = j - 512;
        int f = jj & (HH - 1);
        int o = g * HH + f;
        if (jj < HH)
            v = (RA[o] + 2.0f * RA[3 * S + o]) / c;        // avg(gcn1)+avg(g2)+avg(g2)
        else
            v = funmap(RM[o]) + funmap(RM[3 * S + o]) + funmap(RM[4 * S + o]);
    }
    M[idx] = v;
}

__global__ void fc1_k(const float* __restrict__ M, const float* __restrict__ W,
                      const float* __restrict__ b, float* __restrict__ F1) {
    int idx = blockIdx.x * blockDim.x + threadIdx.x;       // GG*128
    if (idx >= GG * 128) return;
    int g = idx >> 7, j = idx & 127;
    const float* m = M + (size_t)g * 1024;
    float s = b[j];
    for (int k = 0; k < 1024; ++k) s += m[k] * W[k * 128 + j];
    F1[idx] = fmaxf(s, 0.0f);
}

__global__ void fc2_k(const float* __restrict__ F1, const float* __restrict__ W,
                      const float* __restrict__ b, float* __restrict__ out) {
    int idx = blockIdx.x * blockDim.x + threadIdx.x;       // GG*2
    if (idx >= GG * 2) return;
    int g = idx >> 1, j = idx & 1;
    const float* f = F1 + (size_t)g * 128;
    float s = b[j];
    for (int k = 0; k < 128; ++k) s += f[k] * W[k * 2 + j];
    out[idx] = 1.0f / (1.0f + __expf(-s));
}

// ---------------- host driver ----------------
extern "C" void kernel_launch(void* const* d_in, const int* in_sizes, int n_in,
                              void* d_out, int out_size, void* d_ws, size_t ws_size,
                              hipStream_t stream) {
    (void)in_sizes; (void)n_in; (void)out_size; (void)ws_size;

    const float* h_in    = (const float*)d_in[0];
    const int*   src     = (const int*)d_in[1];
    const int*   dst     = (const int*)d_in[2];
    const int*   gid     = (const int*)d_in[3];
    const float* gcn_W[3] = { (const float*)d_in[4], (const float*)d_in[6], (const float*)d_in[8] };
    const float* gcn_b[3] = { (const float*)d_in[5], (const float*)d_in[7], (const float*)d_in[9] };
    const float* gat_W[3]  = { (const float*)d_in[10], (const float*)d_in[14], (const float*)d_in[18] };
    const float* gat_al[3] = { (const float*)d_in[11], (const float*)d_in[15], (const float*)d_in[19] };
    const float* gat_ar[3] = { (const float*)d_in[12], (const float*)d_in[16], (const float*)d_in[20] };
    const float* gat_b[3]  = { (const float*)d_in[13], (const float*)d_in[17], (const float*)d_in[21] };
    const float* fc1_W = (const float*)d_in[22];
    const float* fc1_b = (const float*)d_in[23];
    const float* fc2_W = (const float*)d_in[24];
    const float* fc2_b = (const float*)d_in[25];
    float* out = (float*)d_out;

    // ---- workspace layout (cursor) ----
    char* base = (char*)d_ws;
    size_t off = 0;
    auto alloc = [&](size_t bytes) -> char* {
        off = (off + 255) & ~(size_t)255;
        char* p = base + off;
        off += bytes;
        return p;
    };
    float*    DSRC = (float*)alloc(NN * 4);
    float*    DDST = (float*)alloc(NN * 4);
    float*    CNT  = (float*)alloc(GG * 4);
    __bf16*   Wt_gcn[3];
    Wt_gcn[0] = (__bf16*)alloc((size_t)HH * FIN * 2);
    Wt_gcn[1] = (__bf16*)alloc((size_t)HH * HH * 2);
    Wt_gcn[2] = (__bf16*)alloc((size_t)HH * HH * 2);
    __bf16*   Wt_gat[3];
    Wt_gat[0] = (__bf16*)alloc((size_t)2 * HH * FIN * 2);
    Wt_gat[1] = (__bf16*)alloc((size_t)2 * HH * HH * 2);
    Wt_gat[2] = (__bf16*)alloc((size_t)2 * HH * HH * 2);
    float*    X0   = (float*)alloc((size_t)NN * HH * 4);
    float*    X1   = (float*)alloc((size_t)NN * HH * 4);
    float*    AGG  = (float*)alloc((size_t)NN * HH * 4);
    __bf16*   ABF  = (__bf16*)alloc((size_t)NN * HH * 2);   // bf16 staged GEMM A
    float*    Z    = (float*)alloc((size_t)NN * 2 * HH * 4);
    float*    O    = (float*)alloc((size_t)NN * 2 * HH * 4);
    float*    EL   = (float*)alloc((size_t)NN * 2 * 4);
    float*    ER   = (float*)alloc((size_t)NN * 2 * 4);
    unsigned* EMAX = (unsigned*)alloc((size_t)NN * 2 * 4);
    float*    EDEN = (float*)alloc((size_t)NN * 2 * 4);
    float*    EVAL = (float*)alloc((size_t)EE * 2 * 4);
    float*    RA   = (float*)alloc((size_t)5 * GG * HH * 4);
    unsigned* RM   = (unsigned*)alloc((size_t)5 * GG * HH * 4);
    float*    MRG  = (float*)alloc((size_t)GG * 1024 * 4);
    float*    F1   = (float*)alloc((size_t)GG * 128 * 4);

    const int T = 256;
    auto fillf = [&](float* p, float v, int n) {
        fill_f32<<<CDIV(n, T), T, 0, stream>>>(p, v, n);
    };
    auto fillu = [&](unsigned* p, unsigned v, int n) {
        fill_u32<<<CDIV(n, T), T, 0, stream>>>(p, v, n);
    };

    // ---- phase 0: degrees, counts, weight conversion, accumulator init ----
    fillf(DSRC, 0.0f, NN);
    fillf(DDST, 0.0f, NN);
    fillf(CNT, 0.0f, GG);
    degrees_k<<<CDIV(EE, T), T, 0, stream>>>(src, dst, DSRC, DDST);
    deg_finalize_k<<<CDIV(NN, T), T, 0, stream>>>(DSRC, NN);
    deg_finalize_k<<<CDIV(NN, T), T, 0, stream>>>(DDST, NN);
    counts_k<<<CDIV(NN, T), T, 0, stream>>>(gid, CNT);

    int gcnK[3] = { FIN, HH, HH };
    for (int i = 0; i < 3; ++i)
        wtrans_k<<<CDIV(gcnK[i] * HH, T), T, 0, stream>>>(gcn_W[i], Wt_gcn[i], gcnK[i], HH);
    int gatK[3] = { FIN, HH, HH };
    for (int i = 0; i < 3; ++i)
        wtrans_k<<<CDIV(gatK[i] * 2 * HH, T), T, 0, stream>>>(gat_W[i], Wt_gat[i], gatK[i], 2 * HH);

    fillf(RA, 0.0f, 5 * GG * HH);
    fillu(RM, NEG_INF_MAPPED, 5 * GG * HH);

    auto cvtA = [&](const float* X, const float* rs, int K) {
        int kshift = (K == 128) ? 7 : 8;
        int total4 = NN * K / 4;
        cvt_bf16_k<<<CDIV(total4, T), T, 0, stream>>>(X, rs, ABF, total4, kshift);
    };
    auto gemm = [&](const __bf16* Wt, const float* b, float* C, int K, int Hout, int relu) {
        dim3 grid(CDIV(NN, 16), Hout / 64);
        gemm_wmma_k<<<grid, 128, 0, stream>>>(ABF, Wt, b, C, NN, K, Hout, relu);
    };
    auto readout = [&](const float* X, int slot) {
        readout_k<<<CDIV(NN * HH, T), T, 0, stream>>>(X, gid, RA + slot * GG * HH,
                                                      RM + slot * GG * HH);
    };
    auto gcn_layer = [&](const float* x, int K, int li, float* outbuf) {
        fillf(AGG, 0.0f, NN * K);
        gcn_scatter_k<<<CDIV(EE * (K / 4), T), T, 0, stream>>>(x, src, dst, DSRC, AGG, K);
        cvtA(AGG, DDST, K);                     // fold D_in^{-1/2} row scale
        gemm(Wt_gcn[li], gcn_b[li], outbuf, K, HH, 1);
        readout(outbuf, li);
    };
    auto gat_layer = [&](const float* x, int K, int li, float* outbuf) {
        cvtA(x, nullptr, K);
        gemm(Wt_gat[li], nullptr, Z, K, 2 * HH, 0);
        gat_dots_k<<<CDIV(NN * 2 * 32, 64), 64, 0, stream>>>(Z, gat_al[li], gat_ar[li], EL, ER);
        fillu(EMAX, NEG_INF_MAPPED, NN * 2);
        edge_max_k<<<CDIV(EE * 2, T), T, 0, stream>>>(src, dst, EL, ER, EVAL, EMAX);
        fillf(EDEN, 0.0f, NN * 2);
        edge_exp_k<<<CDIV(EE * 2, T), T, 0, stream>>>(dst, EVAL, EMAX, EDEN);
        fillf(O, 0.0f, NN * 2 * HH);
        edge_aggr_k<<<CDIV(EE * 2 * (HH / 4), T), T, 0, stream>>>(src, dst, Z, EVAL, EDEN, O);
        gat_final_k<<<CDIV(NN * HH, T), T, 0, stream>>>(O, gat_b[li], outbuf);
    };

    // ---- GCN chain ----
    gcn_layer(h_in, FIN, 0, X0);   // gcn1 -> X0, readout slot 0
    gcn_layer(X0,   HH,  1, X1);   // gcn2 -> X1, readout slot 1
    gcn_layer(X1,   HH,  2, X0);   // gcn3 -> X0, readout slot 2

    // ---- GAT chain ----
    gat_layer(h_in, FIN, 0, X1);   // g1 -> X1 (readout unused, faithful)
    gat_layer(X1,   HH,  1, X0);   // g2 -> X0
    readout(X0, 3);                // avg+max of g2
    gat_layer(X0,   HH,  2, X1);   // g3 -> X1
    readout(X1, 4);                // max of g3 (avg unused)

    // ---- merge + MLP head ----
    merged_k<<<CDIV(GG * 1024, T), T, 0, stream>>>(RA, RM, CNT, MRG);
    fc1_k<<<CDIV(GG * 128, T), T, 0, stream>>>(MRG, fc1_W, fc1_b, F1);
    fc2_k<<<CDIV(GG * 2, T), T, 0, stream>>>(F1, fc2_W, fc2_b, out);
}
